// ATConv_2284922601713
// MI455X (gfx1250) — matrix operand order repaired and checked by reading, not compiled, not run on hardware
//
#include <hip/hip_runtime.h>
#include <hip/hip_bf16.h>
#include <math.h>

#define CDIM 192
#define HW   9216
#define WD   96

typedef __attribute__((ext_vector_type(16))) __bf16 v16bf;
typedef __attribute__((ext_vector_type(8)))  __bf16 v8bf;
typedef __attribute__((ext_vector_type(8)))  float  v8f;
typedef __attribute__((ext_vector_type(4)))  float  v4f;

union BF16x16 { v16bf v; v8bf h[2]; };
union BF16x8  { v8bf  v; __bf16 e[8]; };

// Async DMA: copy `bytes` (multiple of blockDim*16) global -> LDS, 16B per lane,
// no VGPR data movement; tracked by ASYNCcnt.
__device__ __forceinline__
void async_copy_to_lds(void* lds_dst, const void* gsrc, int bytes, int t, int nthreads) {
    const unsigned lbase = (unsigned)(size_t)lds_dst;      // LDS offset = addr[31:0]
    const unsigned long long gbase = (unsigned long long)(size_t)gsrc;
    for (int off = t * 16; off < bytes; off += nthreads * 16) {
        unsigned laddr = lbase + (unsigned)off;
        unsigned long long gaddr = gbase + (unsigned long long)off;
        asm volatile("global_load_async_to_lds_b128 %0, %1, off"
                     :: "v"(laddr), "v"(gaddr) : "memory");
    }
}
__device__ __forceinline__ void async_wait_all() {
    asm volatile("s_wait_asynccnt 0" ::: "memory");
}

// ---------------- kernel 0: convert weights to bf16 ----------------
__global__ __launch_bounds__(256)
void k_cvt(const float* __restrict__ Wx, const float* __restrict__ Wp,
           __bf16* __restrict__ Wxb, __bf16* __restrict__ Wpb) {
    int i = blockIdx.x * 256 + threadIdx.x;
    if (i < CDIM * CDIM) {
        Wxb[i] = (__bf16)Wx[i];
        Wpb[i] = (__bf16)Wp[i];
    }
}

// ---------------- kernel 1: chunk-mean pool (1024-px chunks) ----------------
__global__ __launch_bounds__(256)
void k_pool(const float* __restrict__ x, float* __restrict__ xm) {
    __shared__ float sred[256];
    const int bc = blockIdx.x;                 // b*C + c
    const float* xb = x + (size_t)bc * HW;
    const int t = threadIdx.x;
    for (int j = 0; j < 9; ++j) {
        const float* p = xb + j * 1024 + t;
        float s = p[0] + p[256] + p[512] + p[768];
        sred[t] = s;
        __syncthreads();
        for (int off = 128; off > 0; off >>= 1) {
            if (t < off) sred[t] += sred[t + off];
            __syncthreads();
        }
        if (t == 0) xm[(size_t)bc * 9 + j] = sred[0] * (1.0f / 1024.0f);
        __syncthreads();
    }
}

// ---------------- kernel 2: dynamic kernel generation ----------------
// k = gelu(Wk @ xm + bk); g = k @ Wg^T + bg; kfin = g - sigmoid(dc)*mean(g)
__global__ __launch_bounds__(192)
void k_kgen(const float* __restrict__ xm, const float* __restrict__ Wk,
            const float* __restrict__ bk, const float* __restrict__ Wg,
            const float* __restrict__ bg, const float* __restrict__ dc,
            float* __restrict__ kfin) {
    __shared__ float sxm[CDIM * 9];
    const int b = blockIdx.x;
    const int c = threadIdx.x;
    for (int i = c; i < CDIM * 9; i += 192) sxm[i] = xm[(size_t)b * CDIM * 9 + i];
    __syncthreads();

    float kg[9];
    const float* wr = Wk + (size_t)c * CDIM;
    #pragma unroll
    for (int j = 0; j < 9; ++j) {
        float a = bk[c];
        for (int c2 = 0; c2 < CDIM; ++c2) a += wr[c2] * sxm[c2 * 9 + j];
        kg[j] = 0.5f * a * (1.0f + erff(a * 0.7071067811865476f));  // exact GELU
    }
    float g[9];
    float mean = 0.0f;
    #pragma unroll
    for (int j = 0; j < 9; ++j) {
        float a = bg[j];
        #pragma unroll
        for (int i = 0; i < 9; ++i) a += Wg[j * 9 + i] * kg[i];
        g[j] = a;
        mean += a;
    }
    mean *= (1.0f / 9.0f);
    const float factor = 1.0f / (1.0f + expf(-dc[c]));
    float* o = kfin + (size_t)(b * CDIM + c) * 9;
    #pragma unroll
    for (int j = 0; j < 9; ++j) o[j] = g[j] - factor * mean;
}

// ---------------- kernel 3: xp = Wx @ x + bx (bf16 WMMA, bf16 output) ----------------
// A = x tile (M = 128 px), B = Wx^T (N = 192 oc, fed by row-major Wx rows).
// Weights reach LDS via async DMA overlapped with the VALU f32->bf16 x staging.
__global__ __launch_bounds__(256)
void k_gemm1(const float* __restrict__ x, const __bf16* __restrict__ Wxb,
             const float* __restrict__ bx, __bf16* __restrict__ xp) {
    extern __shared__ __align__(16) char smem[];
    __bf16* lW = (__bf16*)smem;                                  // [192 oc][192 c]
    __bf16* lX = (__bf16*)(smem + CDIM * CDIM * sizeof(__bf16)); // [128 px][192 c]
    const int t   = threadIdx.x;
    const int b   = blockIdx.y;
    const int px0 = blockIdx.x * 128;

    // async global->LDS weight stage (72 KB), overlapped with x staging below
    async_copy_to_lds(lW, Wxb, CDIM * CDIM * (int)sizeof(__bf16), t, 256);

    {   // stage x tile transposed, f32 -> bf16 (coalesced across px)
        const int px = t & 127;
        const int c0 = t >> 7;
        const float* xb = x + (size_t)b * CDIM * HW + px0 + px;
        for (int c = c0; c < CDIM; c += 2)
            lX[px * CDIM + c] = (__bf16)xb[(size_t)c * HW];
    }
    async_wait_all();
    __syncthreads();

    const int w = t >> 5, l = t & 31;
    const int lm = l & 15, sel = l >> 4;
    const __bf16* arow = lX + (w * 16 + lm) * CDIM;  // A: row = px

    for (int nt = 0; nt < 12; ++nt) {
        v8f acc = {};
        const __bf16* brow = lW + (nt * 16 + lm) * CDIM;  // B: lane n = oc, k contiguous
        #pragma unroll
        for (int ks = 0; ks < 6; ++ks) {
            const int kb = ks * 32;
            BF16x16 a, bb;
            // A 16-bit 16x32: lanes 0-15: K kb+0..7 / kb+16..23; lanes 16-31: +8
            const __bf16* ap = arow + kb + sel * 8;
            a.h[0] = *(const v8bf*)(ap);
            a.h[1] = *(const v8bf*)(ap + 16);
            // B 16-bit 32x16: lanes 0-15: K kb+0..7 / kb+16..23; lanes 16-31: +8
            const __bf16* bp_ = brow + kb + sel * 8;
            bb.h[0] = *(const v8bf*)(bp_);
            bb.h[1] = *(const v8bf*)(bp_ + 16);
            acc = __builtin_amdgcn_wmma_f32_16x16x32_bf16(
                false, a.v, false, bb.v, (short)0, acc, false, false);
        }
        // D: VGPR r -> M=r (+8 for upper lanes) = consecutive pixels; N=lm = oc
        const int oc = nt * 16 + lm;
        const float bias = bx[oc];
        BF16x8 pk;
        #pragma unroll
        for (int r = 0; r < 8; ++r) pk.e[r] = (__bf16)(acc[r] + bias);
        __bf16* op = xp + (size_t)(b * CDIM + oc) * HW + px0 + w * 16 + sel * 8;
        *(v8bf*)op = pk.v;   // one 16B store per lane per tile
    }
}

// ---------------- kernel 4: fused depthwise 3x3 + (Wp @ y + bp) ----------------
// grid: (H=96 rows, B); 384 threads = 12 waves; wave = (px-tile 0..5, oc-half 0..1),
// loops 6 oc N-tiles. Weights async-DMA'd to LDS while the depthwise runs on VALU.
__global__ __launch_bounds__(384)
void k_dw_gemm2(const __bf16* __restrict__ xp, const float* __restrict__ kfin,
                const __bf16* __restrict__ Wpb, const float* __restrict__ bp,
                float* __restrict__ out) {
    extern __shared__ __align__(16) char smem[];
    __bf16* lW = (__bf16*)smem;                                  // [192 oc][192 c]
    __bf16* lY = (__bf16*)(smem + CDIM * CDIM * sizeof(__bf16)); // [96 px][192 c]
    const int t   = threadIdx.x;
    const int row = blockIdx.x;   // 0..95
    const int b   = blockIdx.y;

    // async global->LDS weight stage (72 KB), overlapped with depthwise below
    async_copy_to_lds(lW, Wpb, CDIM * CDIM * (int)sizeof(__bf16), t, 384);

    // depthwise 3x3 with SAME zero padding; xp (56 MB) is L2-resident (192 MB L2)
    for (int idx = t; idx < CDIM * WD; idx += 384) {
        const int c  = idx / WD;
        const int px = idx % WD;
        const float* kf = kfin + (size_t)(b * CDIM + c) * 9;
        const __bf16* xpc = xp + (size_t)(b * CDIM + c) * HW;
        float acc = 0.0f;
        #pragma unroll
        for (int dy = -1; dy <= 1; ++dy) {
            const int r2 = row + dy;
            if (r2 < 0 || r2 > 95) continue;
            #pragma unroll
            for (int dx = -1; dx <= 1; ++dx) {
                const int c2 = px + dx;
                if (c2 < 0 || c2 > 95) continue;
                acc += kf[(dy + 1) * 3 + (dx + 1)] * (float)xpc[r2 * WD + c2];
            }
        }
        lY[px * CDIM + c] = (__bf16)acc;
    }
    async_wait_all();
    __syncthreads();

    const int w = t >> 5, l = t & 31;
    const int lm = l & 15, sel = l >> 4;
    const int pxt = w % 6;          // px M-tile: 6 x 16 = 96
    const int och = w / 6;          // oc half: tiles 0-5 or 6-11
    const __bf16* arow = lY + (pxt * 16 + lm) * CDIM;

    for (int nt2 = 0; nt2 < 6; ++nt2) {
        const int nt = och * 6 + nt2;
        v8f acc = {};
        const __bf16* brow = lW + (nt * 16 + lm) * CDIM;
        #pragma unroll
        for (int ks = 0; ks < 6; ++ks) {
            const int kb = ks * 32;
            BF16x16 a, bb;
            const __bf16* ap = arow + kb + sel * 8;
            a.h[0] = *(const v8bf*)(ap);
            a.h[1] = *(const v8bf*)(ap + 16);
            const __bf16* bp_ = brow + kb + sel * 8;
            bb.h[0] = *(const v8bf*)(bp_);
            bb.h[1] = *(const v8bf*)(bp_ + 16);
            acc = __builtin_amdgcn_wmma_f32_16x16x32_bf16(
                false, a.v, false, bb.v, (short)0, acc, false, false);
        }
        const int oc = nt * 16 + lm;
        const float bias = bp[oc];
        float* op = out + (size_t)(b * CDIM + oc) * HW + row * WD + pxt * 16 + sel * 8;
        v4f lo = {acc[0] + bias, acc[1] + bias, acc[2] + bias, acc[3] + bias};
        v4f hi = {acc[4] + bias, acc[5] + bias, acc[6] + bias, acc[7] + bias};
        *(v4f*)(op)     = lo;
        *(v4f*)(op + 4) = hi;
    }
}

// ---------------- launch ----------------
extern "C" void kernel_launch(void* const* d_in, const int* in_sizes, int n_in,
                              void* d_out, int out_size, void* d_ws, size_t ws_size,
                              hipStream_t stream) {
    (void)in_sizes; (void)n_in; (void)out_size; (void)ws_size;
    const float* x  = (const float*)d_in[0];
    const float* Wk = (const float*)d_in[1];
    const float* bk = (const float*)d_in[2];
    const float* Wg = (const float*)d_in[3];
    const float* bg = (const float*)d_in[4];
    const float* Wx = (const float*)d_in[5];
    const float* bx = (const float*)d_in[6];
    const float* Wp = (const float*)d_in[7];
    const float* bp = (const float*)d_in[8];
    const float* dc = (const float*)d_in[9];
    float* out = (float*)d_out;

    char* ws = (char*)d_ws;
    float*  xm   = (float*)(ws);                      // 16*192*9 f32   = 110592 B
    float*  kfin = (float*)(ws + 110592);             // 16*192*9 f32   = 110592 B
    __bf16* Wxb  = (__bf16*)(ws + 221184);            // 192*192 bf16   =  73728 B
    __bf16* Wpb  = (__bf16*)(ws + 294912);            // 192*192 bf16   =  73728 B
    __bf16* xpb  = (__bf16*)(ws + 368640);            // 16*192*9216 bf16 = 56.6 MB

    k_cvt <<<144, 256, 0, stream>>>(Wx, Wp, Wxb, Wpb);
    k_pool<<<16 * CDIM, 256, 0, stream>>>(x, xm);
    k_kgen<<<16, 192, 0, stream>>>(xm, Wk, bk, Wg, bg, dc, kfin);

    const size_t lds1 = (size_t)(CDIM * CDIM + 128 * CDIM) * sizeof(__bf16); // 120 KB
    k_gemm1<<<dim3(72, 16), 256, lds1, stream>>>(x, Wxb, bx, xpb);

    const size_t lds2 = (size_t)(CDIM * CDIM + WD * CDIM) * sizeof(__bf16);  // 108 KB
    k_dw_gemm2<<<dim3(WD, 16), 384, lds2, stream>>>(xpb, kfin, Wpb, bp, out);
}
